// FeatureGRU_16527034155541
// MI455X (gfx1250) — compile-verified, matrix operand order checked by reading, not run on hardware
//
#include <hip/hip_runtime.h>
#include <hip/hip_fp16.h>
#include <math.h>

typedef __attribute__((ext_vector_type(16))) _Float16 v16h;
typedef __attribute__((ext_vector_type(2)))  _Float16 v2h;
typedef __attribute__((ext_vector_type(8)))  float    v8f;
typedef __attribute__((ext_vector_type(4)))  float    v4f;
typedef __attribute__((ext_vector_type(4)))  unsigned int v4u;

#define B_   8
#define C_   256
#define D_   128
#define H_   128
#define W_   128
#define HW_  (H_*W_)            // 16384
#define NPIX (B_*HW_)           // 131072

// ---------------------------------------------------------------------------
// WMMA fragment helpers (v_wmma_f32_16x16x32_f16).
// 16-bit A (16x32, MxK): lane<16 holds row M=lane, halves 0..7 = K0..7,
// halves 8..15 = K16..23; lane>=16 holds same row, K8..15 / K24..31.
// B (32x16, KxN) mirrors this with N in place of M; since B column d of
// (X @ W^T) is W row d (contiguous in K), one loader serves A and B.
// Caller passes base = row_base + ((lane>=16)?8:0); 'off' must be a
// compile-time constant so it folds into the b128 instruction offset.
// ---------------------------------------------------------------------------
union Frag {
    v16h v;
    v4u  q[2];
};

__device__ __forceinline__ void frag_load(Frag& f, const _Float16* base, int off) {
    f.q[0] = *(const v4u*)(base + off);
    f.q[1] = *(const v4u*)(base + off + 16);
}

__device__ __forceinline__ void frag_zero(Frag& f) {
    v4u z = {0u, 0u, 0u, 0u};
    f.q[0] = z; f.q[1] = z;
}

__device__ __forceinline__ v8f wmma16(const Frag& a, const Frag& b, v8f c) {
    return __builtin_amdgcn_wmma_f32_16x16x32_f16(
        false, a.v, false, b.v, (short)0, c, false, false);
}

// ---------------------------------------------------------------------------
__global__ void zero_stats(float* stats) {
    if (threadIdx.x < 128) stats[threadIdx.x] = 0.0f;
}

// ---------------------------------------------------------------------------
// Weight prep: f32 -> f16; conv weights tap-major [9][Cout][Cin] so a WMMA B
// row (output channel) is contiguous along K (input channel).
// ---------------------------------------------------------------------------
__global__ void prep_weights(const float* __restrict__ w_ih,   // [384,384]
                             const float* __restrict__ w_hh,   // [384,128]
                             const float* __restrict__ w1,     // [128,256,3,3]
                             const float* __restrict__ w2,     // [256,128,3,3]
                             _Float16* __restrict__ wih16,
                             _Float16* __restrict__ whh16,
                             _Float16* __restrict__ w1t,       // [9][128][256]
                             _Float16* __restrict__ w2t) {     // [9][256][128]
    const int stride = gridDim.x * blockDim.x;
    int tid = blockIdx.x * blockDim.x + threadIdx.x;
    for (int i = tid; i < 384 * 384; i += stride) wih16[i] = (_Float16)w_ih[i];
    for (int i = tid; i < 384 * 128; i += stride) whh16[i] = (_Float16)w_hh[i];
    for (int i = tid; i < 9 * D_ * C_; i += stride) {
        int tap = i / (D_ * C_);
        int r   = i - tap * (D_ * C_);
        int d   = r / C_, c = r - d * C_;
        w1t[i] = (_Float16)w1[(d * C_ + c) * 9 + tap];
    }
    for (int i = tid; i < 9 * C_ * D_; i += stride) {
        int tap = i / (C_ * D_);
        int r   = i - tap * (C_ * D_);
        int co  = r / D_, ci = r - co * D_;
        w2t[i] = (_Float16)w2[(co * D_ + ci) * 9 + tap];
    }
}

// ---------------------------------------------------------------------------
// Tiled NCHW f32 -> channels-last f16 transpose (64x64 tiles through LDS).
// grid = (HW/64, Cc/64, B), block = 256.
// ---------------------------------------------------------------------------
__global__ void to_cl_f16_tiled(const float* __restrict__ src, _Float16* __restrict__ dst,
                                int Cc, int dst_stride, int dst_off) {
    __shared__ _Float16 tile[64][64 + 8];
    const int p0 = blockIdx.x * 64, c0 = blockIdx.y * 64, b = blockIdx.z;
    const int tj = threadIdx.x & 63;
    const int ti = threadIdx.x >> 6;     // 0..3
    #pragma unroll
    for (int i = ti; i < 64; i += 4)
        tile[i][tj] = (_Float16)src[((long)(b * Cc + c0 + i)) * HW_ + p0 + tj];
    __syncthreads();
    #pragma unroll
    for (int i = ti; i < 64; i += 4)
        dst[((long)(b * HW_ + p0 + i)) * dst_stride + dst_off + c0 + tj] = tile[tj][i];
}

// ---------------------------------------------------------------------------
// Coord MLP (4 -> 64 relu -> 128) -> Xf slice [128:256), broadcast over B.
// ---------------------------------------------------------------------------
__global__ void coord_embed(const float* __restrict__ w1, const float* __restrict__ b1,
                            const float* __restrict__ w2, const float* __restrict__ b2,
                            _Float16* __restrict__ Xf) {
    int pix = blockIdx.x;
    int t   = threadIdx.x;                // 0..127
    int y = pix / W_, x = pix - y * W_;
    float xc = -1.0f + 2.0f * (float)x / (float)(W_ - 1);
    float yc = -1.0f + 2.0f * (float)y / (float)(H_ - 1);
    float c0 = xc, c1 = yc, c2 = xc * (float)H_, c3 = yc * (float)W_;
    __shared__ float hid[64];
    if (t < 64) {
        float a = w1[t * 4 + 0] * c0 + w1[t * 4 + 1] * c1 +
                  w1[t * 4 + 2] * c2 + w1[t * 4 + 3] * c3 + b1[t];
        hid[t] = fmaxf(a, 0.0f);
    }
    __syncthreads();
    float acc = b2[t];
    #pragma unroll 8
    for (int k = 0; k < 64; k++) acc += w2[t * 64 + k] * hid[k];
    _Float16 hv = (_Float16)acc;
    for (int b = 0; b < B_; b++)
        Xf[((long)(b * HW_ + pix)) * 384 + 128 + t] = hv;
}

// ---------------------------------------------------------------------------
// 3x3 SAME conv as implicit GEMM via WMMA. 4 waves/WG, each wave = 16-pixel
// M-tile x 128 cols (8 C tiles). K = 9 taps x CIN, fully unrolled per tap so
// every fragment offset is an instruction-immediate from one base pointer.
// FUSE_GN: accumulate per-(b,group) sum/sumsq (group == N-tile index).
// Epilogue stages the 128x64 f32 tile in LDS, then writes coalesced either
// channels-last f32 (conv1) or NCHW f32 + bias + residual (conv2).
// ---------------------------------------------------------------------------
template<int CIN, int COUT, bool FUSE_GN, bool NCHW_OUT>
__global__ __launch_bounds__(128)
void conv3x3_wmma(const _Float16* __restrict__ acts,   // [B,H,W,CIN] f16
                  const _Float16* __restrict__ wt,     // [9][COUT][CIN] f16
                  const float* __restrict__ bias,      // [COUT]
                  const float* __restrict__ residual,  // NCHW f32 (NCHW_OUT)
                  float* __restrict__ out_cl,          // [N,COUT] f32 (!NCHW_OUT)
                  float* __restrict__ out_nchw,        // NCHW f32 (NCHW_OUT)
                  float* __restrict__ stats) {         // [64][2] (FUSE_GN)
    __shared__ float etile[128][65];                   // [col][pixel]
    __shared__ float sh_s[8], sh_ss[8];

    const int wave  = threadIdx.x >> 5;
    const int lane  = threadIdx.x & 31;
    const int lm    = lane & 15;
    const int koff  = (lane >= 16) ? 8 : 0;
    const int rbase = (lane >= 16) ? 8 : 0;

    const int mbase0 = blockIdx.x * 64;               // 64 pixels, same (b,y)
    const int mbase  = mbase0 + wave * 16;
    const int b      = mbase0 >> 14;
    const int rem0   = mbase0 & (HW_ - 1);
    const int y      = rem0 >> 7;
    const int x0     = (rem0 & (W_ - 1)) + wave * 16;
    const int n_base = blockIdx.y * 128;

    if (threadIdx.x < 8) { sh_s[threadIdx.x] = 0.0f; sh_ss[threadIdx.x] = 0.0f; }
    __syncthreads();

    v8f zero = {};
    v8f acc[8];
    #pragma unroll
    for (int i = 0; i < 8; i++) acc[i] = zero;

    // One base pointer for all B fragments of this lane (col n_base+lm).
    const _Float16* bbase = wt + (long)(n_base + lm) * CIN + koff;

    for (int tap = 0; tap < 9; tap++) {
        const int dy = tap / 3 - 1, dx = tap % 3 - 1;
        const int ys = y + dy;
        if (ys < 0 || ys >= H_) continue;
        const int xs = x0 + lm + dx;
        const bool xvalid = (xs >= 0) && (xs < W_);
        const _Float16* arow = acts + ((long)((b * H_ + ys) * W_ + xs)) * CIN + koff;
        const _Float16* bt   = bbase + tap * COUT * CIN;
        #pragma unroll
        for (int kc = 0; kc < CIN; kc += 32) {
            Frag a;
            if (xvalid) frag_load(a, arow, kc);
            else        frag_zero(a);
            Frag bf[8];
            #pragma unroll
            for (int nt = 0; nt < 8; nt++) frag_load(bf[nt], bt, nt * 16 * CIN + kc);
            #pragma unroll
            for (int nt = 0; nt < 8; nt++) acc[nt] = wmma16(a, bf[nt], acc[nt]);
        }
    }

    // Stage into LDS (+bias); GN partial sums on the way.
    #pragma unroll
    for (int nt = 0; nt < 8; nt++) {
        const int cl = nt * 16 + lm;
        const float bia = bias[n_base + cl];
        float s = 0.0f, ss = 0.0f;
        #pragma unroll
        for (int r = 0; r < 8; r++) {
            float v = acc[nt][r] + bia;
            etile[cl][wave * 16 + rbase + r] = v;
            if (FUSE_GN) { s += v; ss += v * v; }
        }
        if (FUSE_GN) { atomicAdd(&sh_s[nt], s); atomicAdd(&sh_ss[nt], ss); }
    }
    __syncthreads();

    if (FUSE_GN && threadIdx.x < 8) {
        atomicAdd(&stats[(b * 8 + (int)threadIdx.x) * 2 + 0], sh_s[threadIdx.x]);
        atomicAdd(&stats[(b * 8 + (int)threadIdx.x) * 2 + 1], sh_ss[threadIdx.x]);
    }

    if constexpr (NCHW_OUT) {
        // 128 cols x 16 float4 of pixels, coalesced NCHW + residual.
        for (int q = threadIdx.x; q < 128 * 16; q += 128) {
            int cl = q >> 4, pl4 = (q & 15) * 4;
            long oi = ((long)(b * COUT + n_base + cl)) * HW_ + rem0 + pl4;
            v4f rsd = *(const v4f*)&residual[oi];
            v4f v;
            #pragma unroll
            for (int k = 0; k < 4; k++) v[k] = etile[cl][pl4 + k] + rsd[k];
            *(v4f*)&out_nchw[oi] = v;
        }
    } else {
        // 64 pixels x 32 float4 of channels, coalesced channels-last.
        for (int q = threadIdx.x; q < 64 * 32; q += 128) {
            int pl = q >> 5, c4 = (q & 31) * 4;
            v4f v;
            #pragma unroll
            for (int k = 0; k < 4; k++) v[k] = etile[c4 + k][pl];
            *(v4f*)&out_cl[((long)(mbase0 + pl)) * COUT + c4] = v;
        }
    }
}

// ---------------------------------------------------------------------------
// GroupNorm apply + exact GELU, f16 into Xf slice [0:128) (coalesced).
// ---------------------------------------------------------------------------
__global__ void gn_apply(const float* __restrict__ act1, const float* __restrict__ stats,
                         const float* __restrict__ scale, const float* __restrict__ bias,
                         _Float16* __restrict__ Xf) {
    long idx = (long)blockIdx.x * blockDim.x + threadIdx.x;   // N*128
    if (idx >= (long)NPIX * D_) return;
    int  d = (int)(idx & 127);
    long p = idx >> 7;
    int  b = (int)(p >> 14);
    int  g = d >> 4;
    const float cnt = 16.0f * (float)HW_;
    float mu  = stats[(b * 8 + g) * 2 + 0] / cnt;
    float var = stats[(b * 8 + g) * 2 + 1] / cnt - mu * mu;
    float xn  = (act1[idx] - mu) * rsqrtf(var + 1e-5f);
    float v   = xn * scale[d] + bias[d];
    float gel = 0.5f * v * (1.0f + erff(v * 0.70710678118654752f));
    Xf[p * 384 + d] = (_Float16)gel;
}

// ---------------------------------------------------------------------------
// Fused GRU: gi = Xf[N,384] @ w_ih^T (K=384), gh = Xf[:,256:384] @ w_hh^T
// (hf is exactly the prev_hidden slice of Xf, so one A stream feeds both).
// Gates in the epilogue; h staged in LDS then written coalesced as both
// f16 channels-last (conv2 input) and f32 NCHW new_hidden (output 2).
// ---------------------------------------------------------------------------
__global__ __launch_bounds__(128)
void gru_wmma(const _Float16* __restrict__ Xf,     // [N,384] f16
              const _Float16* __restrict__ wih,    // [384,384] f16
              const _Float16* __restrict__ whh,    // [384,128] f16
              const float* __restrict__ b_ih,      // [384]
              const float* __restrict__ b_hh,      // [384]
              const float* __restrict__ prev_h,    // [B,128,H,W] f32
              _Float16* __restrict__ h_cl,         // [N,128] f16
              float* __restrict__ new_hidden) {    // [B,128,H,W] f32
    __shared__ float etile[128][65];               // [d][pixel]

    const int wave  = threadIdx.x >> 5;
    const int lane  = threadIdx.x & 31;
    const int lm    = lane & 15;
    const int koff  = (lane >= 16) ? 8 : 0;
    const int rbase = (lane >= 16) ? 8 : 0;

    const int mbase0 = blockIdx.x * 64;
    const int mbase  = mbase0 + wave * 16;
    const int b      = mbase0 >> 14;
    const int rem0   = mbase0 & (HW_ - 1);
    const int rem    = rem0 + wave * 16;

    const _Float16* ap   = Xf  + (long)(mbase + lm) * 384 + koff;
    const _Float16* wihp = wih + (long)lm * 384 + koff;
    const _Float16* whhp = whh + (long)lm * 128 + koff;

    v8f zero = {};
    for (int dt = 0; dt < 8; dt++) {
        const int d0 = dt * 16;
        const _Float16* wb = wihp + d0 * 384;
        const _Float16* hb = whhp + d0 * 128;
        v8f gi_r = zero, gi_z = zero, gi_n = zero;
        v8f gh_r = zero, gh_z = zero, gh_n = zero;
        #pragma unroll
        for (int kc = 0; kc < 384; kc += 32) {
            Frag a; frag_load(a, ap, kc);
            Frag br, bz, bn;
            frag_load(br, wb, kc);
            frag_load(bz, wb, 128 * 384 + kc);
            frag_load(bn, wb, 256 * 384 + kc);
            gi_r = wmma16(a, br, gi_r);
            gi_z = wmma16(a, bz, gi_z);
            gi_n = wmma16(a, bn, gi_n);
            if (kc >= 256) {                       // hf = Xf[:,256:384]
                Frag hr, hz, hn;
                frag_load(hr, hb, kc - 256);
                frag_load(hz, hb, 128 * 128 + kc - 256);
                frag_load(hn, hb, 256 * 128 + kc - 256);
                gh_r = wmma16(a, hr, gh_r);
                gh_z = wmma16(a, hz, gh_z);
                gh_n = wmma16(a, hn, gh_n);
            }
        }
        // Gates for this 16-column d tile -> LDS stage.
        const int d = d0 + lm;
        const float bir = b_ih[d], biz = b_ih[d + 128], bin = b_ih[d + 256];
        const float bhr = b_hh[d], bhz = b_hh[d + 128], bhn = b_hh[d + 256];
        #pragma unroll
        for (int r = 0; r < 8; r++) {
            float ir = gi_r[r] + bir, iz = gi_z[r] + biz, in_ = gi_n[r] + bin;
            float hr = gh_r[r] + bhr, hz = gh_z[r] + bhz, hn  = gh_n[r] + bhn;
            float rr = 1.0f / (1.0f + expf(-(ir + hr)));
            float zz = 1.0f / (1.0f + expf(-(iz + hz)));
            float nn = tanhf(in_ + rr * hn);
            float hprev = prev_h[((long)(b * D_ + d)) * HW_ + (rem + rbase + r)];
            etile[d][wave * 16 + rbase + r] = (1.0f - zz) * nn + zz * hprev;
        }
    }
    __syncthreads();

    // f16 channels-last h (paired stores, coalesced).
    for (int q = threadIdx.x; q < 64 * 64; q += 128) {
        int pl = q >> 6, c2 = (q & 63) * 2;
        v2h hv;
        hv[0] = (_Float16)etile[c2 + 0][pl];
        hv[1] = (_Float16)etile[c2 + 1][pl];
        *(v2h*)&h_cl[((long)(mbase0 + pl)) * D_ + c2] = hv;
    }
    // f32 NCHW new_hidden (b128 stores, coalesced).
    for (int q = threadIdx.x; q < 128 * 16; q += 128) {
        int d = q >> 4, pl4 = (q & 15) * 4;
        v4f v;
        #pragma unroll
        for (int k = 0; k < 4; k++) v[k] = etile[d][pl4 + k];
        *(v4f*)&new_hidden[((long)(b * D_ + d)) * HW_ + rem0 + pl4] = v;
    }
}

// ---------------------------------------------------------------------------
// kernel_launch
// ---------------------------------------------------------------------------
extern "C" void kernel_launch(void* const* d_in, const int* in_sizes, int n_in,
                              void* d_out, int out_size, void* d_ws, size_t ws_size,
                              hipStream_t stream) {
    const float* current_feat = (const float*)d_in[0];
    const float* prev_hidden  = (const float*)d_in[1];
    const float* conv1_w      = (const float*)d_in[2];
    const float* conv1_b      = (const float*)d_in[3];
    const float* gn_scale     = (const float*)d_in[4];
    const float* gn_bias      = (const float*)d_in[5];
    const float* ce_w1        = (const float*)d_in[6];
    const float* ce_b1        = (const float*)d_in[7];
    const float* ce_w2        = (const float*)d_in[8];
    const float* ce_b2        = (const float*)d_in[9];
    const float* w_ih         = (const float*)d_in[10];
    const float* b_ih         = (const float*)d_in[11];
    const float* w_hh         = (const float*)d_in[12];
    const float* b_hh         = (const float*)d_in[13];
    const float* conv2_w      = (const float*)d_in[14];
    const float* conv2_b      = (const float*)d_in[15];

    float* out_feat   = (float*)d_out;                       // [8,256,128,128]
    float* out_hidden = out_feat + (long)B_ * C_ * HW_;      // [8,128,128,128]

    char* ws = (char*)d_ws;
    size_t o = 0;
    _Float16* cfeat_cl = (_Float16*)(ws + o); o += (size_t)B_ * HW_ * C_ * 2;   //  67.1 MB
    _Float16* Xf       = (_Float16*)(ws + o); o += (size_t)NPIX * 384 * 2;      // 100.7 MB
    float*    act1     = (float*)   (ws + o); o += (size_t)NPIX * D_ * 4;       //  67.1 MB
    _Float16* h_cl     = (_Float16*)(ws + o); o += (size_t)NPIX * D_ * 2;       //  33.6 MB
    _Float16* wih16    = (_Float16*)(ws + o); o += (size_t)384 * 384 * 2;
    _Float16* whh16    = (_Float16*)(ws + o); o += (size_t)384 * 128 * 2;
    _Float16* w1t      = (_Float16*)(ws + o); o += (size_t)9 * D_ * C_ * 2;
    _Float16* w2t      = (_Float16*)(ws + o); o += (size_t)9 * C_ * D_ * 2;
    float*    stats    = (float*)   (ws + o); o += 512;

    zero_stats<<<1, 128, 0, stream>>>(stats);
    prep_weights<<<1024, 256, 0, stream>>>(w_ih, w_hh, conv1_w, conv2_w,
                                           wih16, whh16, w1t, w2t);

    // Channels-last f16 activations (tiled transpose).
    to_cl_f16_tiled<<<dim3(HW_ / 64, C_ / 64, B_), 256, 0, stream>>>(
        current_feat, cfeat_cl, C_, C_, 0);
    to_cl_f16_tiled<<<dim3(HW_ / 64, D_ / 64, B_), 256, 0, stream>>>(
        prev_hidden, Xf, D_, 384, 256);

    // Coord MLP -> Xf[:,128:256].
    coord_embed<<<HW_, 128, 0, stream>>>(ce_w1, ce_b1, ce_w2, ce_b2, Xf);

    // conv1 (WMMA) -> act1 f32 channels-last, fused GN stats.
    conv3x3_wmma<C_, D_, true, false><<<dim3(NPIX / 64, 1), 128, 0, stream>>>(
        cfeat_cl, w1t, conv1_b, nullptr, act1, nullptr, stats);

    // GN apply + GELU -> Xf[:,0:128].
    gn_apply<<<(unsigned)(((long)NPIX * D_) / 256), 256, 0, stream>>>(
        act1, stats, gn_scale, gn_bias, Xf);

    // Fused GRU GEMMs + gates (WMMA) -> h_cl f16 + new_hidden f32 (output 2).
    gru_wmma<<<NPIX / 64, 128, 0, stream>>>(
        Xf, wih16, whh16, b_ih, b_hh, prev_hidden, h_cl, out_hidden);

    // conv2 (WMMA) + bias + residual -> output_feat f32 NCHW (output 1).
    conv3x3_wmma<D_, C_, false, true><<<dim3(NPIX / 64, 2), 128, 0, stream>>>(
        h_cl, w2t, conv2_b, current_feat, nullptr, out_feat, nullptr);
}
